// FeatureLoss_14688788152846
// MI455X (gfx1250) — matrix operand order repaired and checked by reading, not compiled, not run on hardware
//
#include <hip/hip_runtime.h>
#include <hip/hip_bf16.h>
#include <stdint.h>

#define FEAT   4096
#define BATCH  512
#define KROWS  8192            // 2*FEAT contrastive samples
#define TEMP_INV 2.0f          // 1/temperature
#define NCHUNK 8               // j-range splits (deterministic partials, no atomics)
#define WAVES_PER_BLOCK 8
#define ENT_BLOCKS 1024
#define LDSROW 520             // halfs per LDS tile row: 512 + 8 pad (bank decorrelation)

typedef __attribute__((ext_vector_type(16))) _Float16 v16h;
typedef __attribute__((ext_vector_type(8)))  float    v8f;
typedef __attribute__((ext_vector_type(8)))  uint32_t v8u;
typedef __attribute__((ext_vector_type(4)))  int      v4i;

union FragU { v16h h; v8u u; };

#if __has_builtin(__builtin_amdgcn_global_load_async_to_lds_b128)
#define HAVE_ASYNC_LDS 1
typedef __attribute__((address_space(1))) v4i* gv4i_p;
typedef __attribute__((address_space(3))) v4i* lv4i_p;

__device__ __forceinline__ void wait_async0() {
#if __has_builtin(__builtin_amdgcn_s_wait_asynccnt)
    __builtin_amdgcn_s_wait_asynccnt(0);
#else
    asm volatile("s_wait_asynccnt 0x0" ::: "memory");
#endif
}

// Issue this thread's 64-byte share of a 16x512-half B tile as 4 async
// global->LDS b128 copies (no VGPR payload; tracked by ASYNCcnt).
// AS1 pointer = flat address; AS3 pointer = addr[31:0] (wave LDS offset).
__device__ __forceinline__ void async_tile_load(const _Float16* gbase, _Float16* lbase) {
#pragma unroll
    for (int i = 0; i < 4; ++i)
        __builtin_amdgcn_global_load_async_to_lds_b128(
            (gv4i_p)(uintptr_t)(gbase + i * 8),
            (lv4i_p)(uint32_t)(uintptr_t)(lbase + i * 8), 0, 0);
}
#endif

// Load one 32-K-step B fragment (8 dwords) from the LDS tile row pointer.
#define LOAD_BFRAG(dst, ksi)                                        \
    do {                                                            \
        const uint32_t* _p = lb + (ksi) * 16 + h * 8;               \
        _Pragma("unroll")                                           \
        for (int _v = 0; _v < 8; ++_v) (dst).u[_v] = _p[(_v & 3) - h * 4 + h * 4 * 2 - h * 8 + h * 4 + (_v >> 2) * 8 - h * 4]; \
    } while (0)

// ---------------------------------------------------------------------------
// Kernel 1: per-column L2 norm of [f1 | f2] and write normalized f16 matrix
// fn16[row=feature index 0..8191][k=batch 0..511], row-major (K contiguous).
// One wave per column.
// ---------------------------------------------------------------------------
__global__ __launch_bounds__(256)
void prep_normalize(const float* __restrict__ f1, const float* __restrict__ f2,
                    _Float16* __restrict__ fn16) {
    const int lane = threadIdx.x & 31;
    const int wave = threadIdx.x >> 5;
    const int col  = blockIdx.x * (blockDim.x >> 5) + wave;   // 0..8191
    const float* src = (col < FEAT) ? (f1 + col) : (f2 + (col - FEAT));

    float v[16];
    float s = 0.0f;
#pragma unroll
    for (int it = 0; it < 16; ++it) {
        const int b = lane + 32 * it;
        const float x = src[(size_t)b * FEAT];
        v[it] = x;
        s += x * x;
    }
#pragma unroll
    for (int off = 16; off; off >>= 1) s += __shfl_xor(s, off, 32);

    const float inv = 1.0f / fmaxf(sqrtf(s), 1e-8f);
    _Float16* dst = fn16 + (size_t)col * BATCH;
#pragma unroll
    for (int it = 0; it < 16; ++it) {
        const int b = lane + 32 * it;
        dst[b] = (_Float16)(v[it] * inv);
    }
}

// ---------------------------------------------------------------------------
// Kernel 2: entropy partial sums over f1 and f2 (deterministic block tree).
// ---------------------------------------------------------------------------
__global__ __launch_bounds__(256)
void entropy_partial(const float* __restrict__ f1, const float* __restrict__ f2,
                     float* __restrict__ entPart) {
    __shared__ float red[256];
    const int NTOT   = FEAT * BATCH;
    const int idx    = blockIdx.x * blockDim.x + threadIdx.x;
    const int stride = gridDim.x * blockDim.x;
    float s = 0.0f;
    for (int i = idx; i < NTOT; i += stride) {
        float p = f1[i];
        s += -p * logf(p + 1e-12f) - (1.0f - p) * logf(1.0f - p + 1e-12f);
        p = f2[i];
        s += -p * logf(p + 1e-12f) - (1.0f - p) * logf(1.0f - p + 1e-12f);
    }
    red[threadIdx.x] = s;
    __syncthreads();
    for (int off = 128; off; off >>= 1) {
        if ((int)threadIdx.x < off) red[threadIdx.x] += red[threadIdx.x + off];
        __syncthreads();
    }
    if (threadIdx.x == 0) entPart[blockIdx.x] = red[0];
}

// ---------------------------------------------------------------------------
// Kernel 3: self-sim (diag) and positive-pair sim, fp32 from fn16.
// One wave per row.
// ---------------------------------------------------------------------------
__global__ __launch_bounds__(256)
void diag_pos(const _Float16* __restrict__ fn16,
              float* __restrict__ diag, float* __restrict__ pos) {
    const int lane = threadIdx.x & 31;
    const int wave = threadIdx.x >> 5;
    const int row  = blockIdx.x * (blockDim.x >> 5) + wave;
    const int partner = (row < FEAT) ? row + FEAT : row - FEAT;
    const _Float16* a = fn16 + (size_t)row * BATCH;
    const _Float16* b = fn16 + (size_t)partner * BATCH;
    float sd = 0.0f, sp = 0.0f;
#pragma unroll
    for (int it = 0; it < 16; ++it) {
        const int k = lane + 32 * it;
        const float x = (float)a[k];
        sd += x * x;
        sp += x * (float)b[k];
    }
#pragma unroll
    for (int off = 16; off; off >>= 1) {
        sd += __shfl_xor(sd, off, 32);
        sp += __shfl_xor(sp, off, 32);
    }
    if (lane == 0) {
        diag[row] = sd * TEMP_INV;
        pos[row]  = sp * TEMP_INV;
    }
}

// ---------------------------------------------------------------------------
// Kernel 4: main WMMA kernel. 8 waves per block, each owning one 16-row
// i-tile; the block sweeps one j-chunk with B tiles staged in LDS.
// Async path: tile t+1 streams global->LDS via GLOBAL_LOAD_ASYNC_TO_LDS_B128
// (zero VGPR payload) while tile t is consumed; s_wait_asynccnt + barrier
// publishes the buffer. Fallback path: synchronous single-buffer staging.
// B fragments are software-pipelined through two register buffers, and dual
// accumulators break the WMMA D->C RAW chain.
//
// Fragment layouts (CDNA5 16-bit):
//   A (16x32): lane L (m=L&15, h=L>>4) holds row ibase+m; dword v -> K pair
//              kbase + (v&3)*2 + 8h + 16*(v>>2)
//   B (32x16): lane L holds column jbase+m; dword v -> K pair kbase + (v&3)*2 + 8h + 16*(v>>2)
// ---------------------------------------------------------------------------
__global__ __launch_bounds__(WAVES_PER_BLOCK * 32)
void simexp_rows(const _Float16* __restrict__ fn16, float* __restrict__ partial) {
#ifdef HAVE_ASYNC_LDS
    __shared__ _Float16 btile[2][16 * LDSROW];   // double buffer, 2 x 16.25 KB
#else
    __shared__ _Float16 btile[1][16 * LDSROW];   // single buffer
#endif

    const int tid   = threadIdx.x;
    const int lane  = tid & 31;
    const int wave  = tid >> 5;
    const int itile = blockIdx.x * WAVES_PER_BLOCK + wave;   // 0..511
    const int chunk = blockIdx.y;                            // 0..NCHUNK-1
    const int ibase = itile * 16;
    const int m = lane & 15;
    const int h = lane >> 4;

    // Preload all 16 K-step A fragments for this i-tile into registers.
    const uint32_t* rowA = (const uint32_t*)(fn16 + (size_t)(ibase + m) * BATCH);
    FragU afr[16];
#pragma unroll
    for (int ks = 0; ks < 16; ++ks) {
#pragma unroll
        for (int v = 0; v < 8; ++v) {
            const int di = ks * 16 + (v & 3) + h * 4 + (v >> 2) * 8;
            afr[ks].u[v] = rowA[di];
        }
    }

    // Cooperative B-tile loader mapping: 256 threads, 64 B each.
    const int lr = tid >> 4;          // tile row 0..15
    const int ls = tid & 15;          // 64-byte segment 0..15

    const int JT  = (KROWS / 16) / NCHUNK;     // 64 j-tiles per chunk
    const int jt0 = chunk * JT;

    v8f acc = {};

#ifdef HAVE_ASYNC_LDS
    // Prologue: stream tile 0 into buffer 0.
    async_tile_load(fn16 + (size_t)(jt0 * 16 + lr) * BATCH + ls * 32,
                    &btile[0][lr * LDSROW + ls * 32]);
    wait_async0();
    __syncthreads();

    for (int jt = 0; jt < JT; ++jt) {
        const int cur = jt & 1;

        // Stream next tile into the idle buffer (everyone passed the previous
        // end-barrier, so no wave is still reading buf[cur^1]).
        if (jt + 1 < JT)
            async_tile_load(fn16 + (size_t)((jt0 + jt + 1) * 16 + lr) * BATCH + ls * 32,
                            &btile[cur ^ 1][lr * LDSROW + ls * 32]);

        // Compute on current buffer: 16 K-steps, pipelined fragments.
        const uint32_t* lb = (const uint32_t*)&btile[cur][(size_t)m * LDSROW];
        v8f d0 = {}, d1 = {};
        FragU fb0, fb1;
        LOAD_BFRAG(fb0, 0);
#pragma unroll
        for (int ks = 0; ks < 16; ks += 2) {
            LOAD_BFRAG(fb1, ks + 1);
            d0 = __builtin_amdgcn_wmma_f32_16x16x32_f16(
                     false, afr[ks].h, false, fb0.h, (short)0, d0, false, false);
            if (ks + 2 < 16) LOAD_BFRAG(fb0, ks + 2);
            d1 = __builtin_amdgcn_wmma_f32_16x16x32_f16(
                     false, afr[ks + 1].h, false, fb1.h, (short)0, d1, false, false);
        }
#pragma unroll
        for (int r = 0; r < 8; ++r) acc[r] += expf(TEMP_INV * (d0[r] + d1[r]));

        if (jt + 1 < JT) {
            wait_async0();     // next buffer fully written (this wave's share)
            __syncthreads();   // all waves' shares written + all reads of cur done
        }
    }
#else
    // Fallback: synchronous single-buffer staging (short-lived regs, no spill).
    for (int jt = 0; jt < JT; ++jt) {
        uint4 g[4];
        const uint4* p =
            (const uint4*)(fn16 + (size_t)((jt0 + jt) * 16 + lr) * BATCH) + ls * 4;
#pragma unroll
        for (int i = 0; i < 4; ++i) g[i] = p[i];
        if (jt) __syncthreads();                   // previous compute finished
        uint4* q = (uint4*)&btile[0][0] + lr * (LDSROW / 8) + ls * 4;
#pragma unroll
        for (int i = 0; i < 4; ++i) q[i] = g[i];
        __syncthreads();

        const uint32_t* lb = (const uint32_t*)&btile[0][(size_t)m * LDSROW];
        v8f d0 = {}, d1 = {};
        FragU fb0, fb1;
        LOAD_BFRAG(fb0, 0);
#pragma unroll
        for (int ks = 0; ks < 16; ks += 2) {
            LOAD_BFRAG(fb1, ks + 1);
            d0 = __builtin_amdgcn_wmma_f32_16x16x32_f16(
                     false, afr[ks].h, false, fb0.h, (short)0, d0, false, false);
            if (ks + 2 < 16) LOAD_BFRAG(fb0, ks + 2);
            d1 = __builtin_amdgcn_wmma_f32_16x16x32_f16(
                     false, afr[ks + 1].h, false, fb1.h, (short)0, d1, false, false);
        }
#pragma unroll
        for (int r = 0; r < 8; ++r) acc[r] += expf(TEMP_INV * (d0[r] + d1[r]));
    }
#endif

    // Reduce each row (8 rows per 16-lane half) across its 16 N-lanes.
#pragma unroll
    for (int r = 0; r < 8; ++r) {
        float s = acc[r];
#pragma unroll
        for (int off = 8; off; off >>= 1) s += __shfl_xor(s, off, 16);
        if (m == 0)
            partial[(size_t)chunk * KROWS + (ibase + r + 8 * h)] = s;
    }
}

// ---------------------------------------------------------------------------
// Kernel 5: deterministic final reduction to the scalar loss.
// ---------------------------------------------------------------------------
__global__ __launch_bounds__(256)
void finalize(const float* __restrict__ partial, const float* __restrict__ diag,
              const float* __restrict__ pos, const float* __restrict__ entPart,
              float* __restrict__ out) {
    __shared__ float redL[256];
    __shared__ float redE[256];
    float sL = 0.0f;
    for (int i = threadIdx.x; i < KROWS; i += 256) {
        float rs = 0.0f;
        for (int c = 0; c < NCHUNK; ++c) rs += partial[(size_t)c * KROWS + i];
        rs -= expf(diag[i]);                 // remove self term (masked diagonal)
        sL += logf(fmaxf(rs, 1e-30f)) - pos[i];
    }
    float sE = 0.0f;
    for (int i = threadIdx.x; i < ENT_BLOCKS; i += 256) sE += entPart[i];
    redL[threadIdx.x] = sL;
    redE[threadIdx.x] = sE;
    __syncthreads();
    for (int off = 128; off; off >>= 1) {
        if ((int)threadIdx.x < off) {
            redL[threadIdx.x] += redL[threadIdx.x + off];
            redE[threadIdx.x] += redE[threadIdx.x + off];
        }
        __syncthreads();
    }
    if (threadIdx.x == 0) {
        const float loss   = redL[0] / (float)KROWS;
        const float neloss = redE[0] / (2.0f * (float)(FEAT * BATCH) * 0.6931471805599453f);
        out[0] = loss - neloss;
    }
}

// ---------------------------------------------------------------------------
extern "C" void kernel_launch(void* const* d_in, const int* in_sizes, int n_in,
                              void* d_out, int out_size, void* d_ws, size_t ws_size,
                              hipStream_t stream) {
    const float* f1 = (const float*)d_in[0];
    const float* f2 = (const float*)d_in[1];
    float* out = (float*)d_out;

    char* ws = (char*)d_ws;
    _Float16* fn16 = (_Float16*)ws;                                   // 8,388,608 B
    float* partial = (float*)(ws + (size_t)KROWS * BATCH * 2);        // 8*8192*4 B
    float* diag    = partial + (size_t)NCHUNK * KROWS;
    float* pos     = diag + KROWS;
    float* entPart = pos + KROWS;

    prep_normalize<<<KROWS / 8, 256, 0, stream>>>(f1, f2, fn16);
    entropy_partial<<<ENT_BLOCKS, 256, 0, stream>>>(f1, f2, entPart);
    diag_pos<<<KROWS / 8, 256, 0, stream>>>(fn16, diag, pos);

    dim3 grid((KROWS / 16) / WAVES_PER_BLOCK, NCHUNK);                // 64 x 8
    simexp_rows<<<grid, WAVES_PER_BLOCK * 32, 0, stream>>>(fn16, partial);

    finalize<<<1, 256, 0, stream>>>(partial, diag, pos, entPart, out);
}
